// Glbp_75780402970626
// MI455X (gfx1250) — matrix operand order repaired and checked by reading, not compiled, not run on hardware
//
#include <hip/hip_runtime.h>
#include <hip/hip_bf16.h>

typedef __attribute__((ext_vector_type(16))) _Float16 v16h;
typedef __attribute__((ext_vector_type(8)))  _Float16 v8h;
typedef __attribute__((ext_vector_type(8)))  float    v8f;

#define NUu 339
#define NIi 5825
#define NN  6164
#define BB  32768
#define SS  20
#define DD  128

// ---------------- WMMA fragment helpers (wave32, v_wmma_f32_16x16x32_f16) ---

// A: 16x32 tile (MxK) from row-major storage, leading dim ld (halves).
// Layout: lane (M=lane&15): halves 0-15 = K hi*16..hi*16+15 (hi = lane>>4),
//         matching packed-B contiguous k order.
// One contiguous run -> two b128 loads per lane.
__device__ __forceinline__ v16h load_a_frag(const _Float16* p, int ld, int lane) {
  const int row = lane & 15;
  const int hi  = lane >> 4;
  const _Float16* rp = p + row * ld + hi * 16;
  const v8h lo = *(const v8h*)(rp);
  const v8h hh = *(const v8h*)(rp + 8);
  v16h a;
#pragma unroll
  for (int i = 0; i < 8; ++i) { a[i] = lo[i]; a[8 + i] = hh[i]; }
  return a;
}

// B: 32x16 tile (KxN) stored in packed fragment order:
// tile_base[lane*16 + j] == B[(lane>>4)*16 + j][lane&15]
// -> one contiguous 32-byte read per lane (2x b128).
__device__ __forceinline__ v16h load_b_packed(const _Float16* tile_base, int lane) {
  const _Float16* rp = tile_base + lane * 16;
  const v8h lo = *(const v8h*)(rp);
  const v8h hh = *(const v8h*)(rp + 8);
  v16h b;
#pragma unroll
  for (int i = 0; i < 8; ++i) { b[i] = lo[i]; b[8 + i] = hh[i]; }
  return b;
}

// Store a 16x16 f32 D fragment into packed-B-fragment-order f16 storage.
// Element (gr = mt*16 + hi*8 + r, gc = nt*16 + col) -> tile nt,
// offset (mt*16+col)*16 + hi*8 + r : contiguous in r -> one b128 store.
__device__ __forceinline__ void store_d_packed(_Float16* tile_base, int mt, int lane,
                                               const v8f& acc) {
  const int col = lane & 15, hi = lane >> 4;
  v8h ov;
#pragma unroll
  for (int r = 0; r < 8; ++r) ov[r] = (_Float16)acc[r];
  *(v8h*)(tile_base + (mt * 16 + col) * 16 + hi * 8) = ov;
}

// ---------------- prep: adj_x = l2norm(info @ W + b), stored f16 -------------

__global__ __launch_bounds__(128) void prep_adjx(
    const float* __restrict__ uinfo, const float* __restrict__ iinfo,
    const float* __restrict__ Wu, const float* __restrict__ bu,
    const float* __restrict__ Wi, const float* __restrict__ bi,
    _Float16* __restrict__ adjx16) {
  const int n = blockIdx.x;
  const int j = threadIdx.x;
  const float* info; const float* W; const float* bias;
  if (n < NUu) { info = uinfo + (size_t)n * 8;          W = Wu; bias = bu; }
  else         { info = iinfo + (size_t)(n - NUu) * 8;  W = Wi; bias = bi; }
  float v = bias[j];
#pragma unroll
  for (int f = 0; f < 8; ++f) v += info[f] * W[f * DD + j];
  __shared__ float red[128];
  red[j] = v * v;
  __syncthreads();
  for (int s = 64; s > 0; s >>= 1) {
    if (j < s) red[j] += red[j + s];
    __syncthreads();
  }
  const float denom = fmaxf(sqrtf(red[0]), 1e-12f);
  adjx16[(size_t)n * DD + j] = (_Float16)(v / denom);
}

// ---------------- weight pack: f32 row-major -> f16 fragment-order -----------
// Packed element [t*512 + lane*16 + j] = W[(kk*32 + (lane>>4)*16 + j)*ldw + nt*16 + (lane&15)]
// W1  (128x256): t = kk*16+nt (kk<4, nt<16)   W2  (256x128): t = kk*8+nt (kk<8, nt<8)
// mW0 (256x128): t = kk*8+nt  (kk<8, nt<8)    mW1 (128x64):  t = kk*4+nt (kk<4, nt<4)

__global__ __launch_bounds__(256) void pack_w(
    const float* __restrict__ W1, const float* __restrict__ W2,
    const float* __restrict__ mW0, const float* __restrict__ mW1,
    _Float16* __restrict__ W1p, _Float16* __restrict__ W2p,
    _Float16* __restrict__ mW0p, _Float16* __restrict__ mW1p) {
  const int idx = blockIdx.x * 256 + threadIdx.x;   // 0..32767
  const int t = idx >> 9, w = idx & 511, lane = w >> 4, j = w & 15;
  const int klane = lane >> 4, col = lane & 15;
  {
    const int kk = t >> 4, nt = t & 15;
    W1p[idx] = (_Float16)W1[(kk * 32 + klane * 16 + j) * 256 + nt * 16 + col];
  }
  {
    const int kk = t >> 3, nt = t & 7;
    W2p[idx]  = (_Float16)W2 [(kk * 32 + klane * 16 + j) * 128 + nt * 16 + col];
    mW0p[idx] = (_Float16)mW0[(kk * 32 + klane * 16 + j) * 128 + nt * 16 + col];
  }
  if (idx < 16 * 512) {
    const int kk = t >> 2, nt = t & 3;
    mW1p[idx] = (_Float16)mW1[(kk * 32 + klane * 16 + j) * 64 + nt * 16 + col];
  }
}

// ---------------- fused per-batch GNN (writes blended x to xbuf) -------------

__global__ __launch_bounds__(256) void gnn_kernel(
    const int* __restrict__ user_idx, const int* __restrict__ item_idx,
    const int* __restrict__ sample_ids, const int* __restrict__ roots,
    const float* __restrict__ adj_matrix,
    const float* __restrict__ uid_emb, const float* __restrict__ sid_emb,
    const _Float16* __restrict__ adjx16,
    const _Float16* __restrict__ W1p, const float* __restrict__ b1,
    const _Float16* __restrict__ W2p, const float* __restrict__ b2,
    _Float16* __restrict__ xbuf) {
  const int b    = blockIdx.x;
  const int tid  = threadIdx.x;
  const int lane = tid & 31;
  const int wv   = tid >> 5;            // 8 wave32s per workgroup

  __shared__ int s_ids[32];
  __shared__ __align__(32) _Float16 s_adj[32 * 32];    // padded sub_adj (row-major)
  __shared__ __align__(32) _Float16 s_x  [32 * 128];   // padded sub_x   (row-major)
  __shared__ __align__(32) _Float16 s_t1p[16 * 512];   // t1, packed B-frag tiles
  __shared__ __align__(32) _Float16 s_h1 [32 * 256];   // h1 (row-major, A source)
  __shared__ __align__(32) _Float16 s_t2p[8 * 512];    // t2, packed B-frag tiles
  __shared__ float s_h2[32 * 128];

  // ---- gather sample ids ----
  if (tid < 32) s_ids[tid] = (tid < SS) ? sample_ids[(size_t)b * SS + tid] : 0;
  __syncthreads();

  // ---- gather sub_adj (32x32, zero padded) ----
  for (int idx = tid; idx < 32 * 32; idx += 256) {
    const int r = idx >> 5, c = idx & 31;
    float v = 0.0f;
    if (r < SS && c < SS) v = adj_matrix[(size_t)s_ids[r] * NN + s_ids[c]];
    s_adj[idx] = (_Float16)v;
  }
  // ---- gather sub_x (32x128, zero padded), 16B vectors ----
  for (int chunk = tid; chunk < 32 * 16; chunk += 256) {
    const int r = chunk >> 4, c8 = (chunk & 15) * 8;
    v8h v = {};
    if (r < SS) v = *(const v8h*)(adjx16 + (size_t)s_ids[r] * DD + c8);
    *(v8h*)(s_x + r * 128 + c8) = v;
  }
  __syncthreads();

  // ---- stage 1: t1 = sub_x(32x128) @ W1(128x256) -> 2x16 tiles, K=4x32 ----
  for (int t = wv; t < 32; t += 8) {
    const int mt = t >> 4, nt = t & 15;
    v8f acc = {};
#pragma unroll
    for (int kk = 0; kk < 4; ++kk) {
      const v16h a  = load_a_frag(s_x + (mt * 16) * 128 + kk * 32, 128, lane);
      const v16h bf = load_b_packed(W1p + (kk * 16 + nt) * 512, lane);
      acc = __builtin_amdgcn_wmma_f32_16x16x32_f16(false, a, false, bf,
                                                   (short)0, acc, false, false);
    }
    store_d_packed(s_t1p + nt * 512, mt, lane, acc);
  }
  __syncthreads();

  // ---- stage 2: h1 = relu(adj(32x32) @ t1(32x256) + b1) ----
  for (int t = wv; t < 32; t += 8) {
    const int mt = t >> 4, nt = t & 15;
    const v16h a  = load_a_frag(s_adj + (mt * 16) * 32, 32, lane);
    const v16h bf = load_b_packed(s_t1p + nt * 512, lane);
    v8f acc = {};
    acc = __builtin_amdgcn_wmma_f32_16x16x32_f16(false, a, false, bf,
                                                 (short)0, acc, false, false);
    const int hi = lane >> 4, col = lane & 15;
    const float bias = b1[nt * 16 + col];
#pragma unroll
    for (int r = 0; r < 8; ++r) {
      const float v = fmaxf(acc[r] + bias, 0.0f);
      s_h1[(mt * 16 + hi * 8 + r) * 256 + nt * 16 + col] = (_Float16)v;
    }
  }
  __syncthreads();

  // ---- stage 3: t2 = h1(32x256) @ W2(256x128) -> 2x8 tiles, K=8x32 ----
  for (int t = wv; t < 16; t += 8) {
    const int mt = t >> 3, nt = t & 7;
    v8f acc = {};
#pragma unroll
    for (int kk = 0; kk < 8; ++kk) {
      const v16h a  = load_a_frag(s_h1 + (mt * 16) * 256 + kk * 32, 256, lane);
      const v16h bf = load_b_packed(W2p + (kk * 8 + nt) * 512, lane);
      acc = __builtin_amdgcn_wmma_f32_16x16x32_f16(false, a, false, bf,
                                                   (short)0, acc, false, false);
    }
    store_d_packed(s_t2p + nt * 512, mt, lane, acc);
  }
  __syncthreads();

  // ---- stage 4: h2 = relu(adj(32x32) @ t2(32x128) + b2), kept f32 ----
  for (int t = wv; t < 16; t += 8) {
    const int mt = t >> 3, nt = t & 7;
    const v16h a  = load_a_frag(s_adj + (mt * 16) * 32, 32, lane);
    const v16h bf = load_b_packed(s_t2p + nt * 512, lane);
    v8f acc = {};
    acc = __builtin_amdgcn_wmma_f32_16x16x32_f16(false, a, false, bf,
                                                 (short)0, acc, false, false);
    const int hi = lane >> 4, col = lane & 15;
    const float bias = b2[nt * 16 + col];
#pragma unroll
    for (int r = 0; r < 8; ++r)
      s_h2[(mt * 16 + hi * 8 + r) * 128 + nt * 16 + col] = fmaxf(acc[r] + bias, 0.0f);
  }
  __syncthreads();

  // ---- combine: x = 0.6*id_vec + 0.4*ft_vec -> xbuf (f16) ----
  {
    const int u  = user_idx[b];
    const int it = item_idx[b];
    const int r0 = roots[2 * b];
    const int r1 = roots[2 * b + 1];
    const int j = tid;
    float idv, ftv;
    if (j < 128) { idv = uid_emb[(size_t)u * DD + j];           ftv = s_h2[r0 * 128 + j]; }
    else         { idv = sid_emb[(size_t)it * DD + (j - 128)];  ftv = s_h2[r1 * 128 + (j - 128)]; }
    xbuf[(size_t)b * 256 + j] = (_Float16)(0.6f * idv + 0.4f * ftv);
  }
}

// ---------------- batched MLP head: WMMA GEMMs over all 32768 rows -----------
// 128 rows per workgroup; wave w owns M-tile w (16 rows).

__global__ __launch_bounds__(256) void mlp_kernel(
    const _Float16* __restrict__ xbuf,
    const _Float16* __restrict__ mW0p, const float* __restrict__ mb0,
    const _Float16* __restrict__ mW1p, const float* __restrict__ mb1,
    const float* __restrict__ mW2, const float* __restrict__ mb2,
    float* __restrict__ out) {
  const int tid  = threadIdx.x;
  const int lane = tid & 31;
  const int wv   = tid >> 5;
  const int row0 = blockIdx.x * 128;

  __shared__ __align__(32) _Float16 s_y1[128 * 128];  // relu(X@mW0+b), f16 row-major
  __shared__ float s_y2[128 * 64];                    // relu(y1@mW1+b), f32 row-major

  const int hi = lane >> 4, col = lane & 15;

  // ---- layer 0: y1 = relu(X(128x256) @ mW0(256x128) + mb0) ----
  const _Float16* xa = xbuf + (size_t)(row0 + wv * 16) * 256;
#pragma unroll
  for (int nt = 0; nt < 8; ++nt) {
    v8f acc = {};
#pragma unroll
    for (int kk = 0; kk < 8; ++kk) {
      const v16h a  = load_a_frag(xa + kk * 32, 256, lane);
      const v16h bf = load_b_packed(mW0p + (kk * 8 + nt) * 512, lane);
      acc = __builtin_amdgcn_wmma_f32_16x16x32_f16(false, a, false, bf,
                                                   (short)0, acc, false, false);
    }
    const float bias = mb0[nt * 16 + col];
#pragma unroll
    for (int r = 0; r < 8; ++r) {
      const float v = fmaxf(acc[r] + bias, 0.0f);
      s_y1[(wv * 16 + hi * 8 + r) * 128 + nt * 16 + col] = (_Float16)v;
    }
  }
  // Wave w's layer-1 A-tiles are exactly the rows it just wrote (DS in-order
  // within a wave), so no workgroup barrier is required here.

  // ---- layer 1: y2 = relu(y1(128x128) @ mW1(128x64) + mb1) ----
#pragma unroll
  for (int nt = 0; nt < 4; ++nt) {
    v8f acc = {};
#pragma unroll
    for (int kk = 0; kk < 4; ++kk) {
      const v16h a  = load_a_frag(s_y1 + (wv * 16) * 128 + kk * 32, 128, lane);
      const v16h bf = load_b_packed(mW1p + (kk * 4 + nt) * 512, lane);
      acc = __builtin_amdgcn_wmma_f32_16x16x32_f16(false, a, false, bf,
                                                   (short)0, acc, false, false);
    }
    const float bias = mb1[nt * 16 + col];
#pragma unroll
    for (int r = 0; r < 8; ++r)
      s_y2[(wv * 16 + hi * 8 + r) * 64 + nt * 16 + col] = fmaxf(acc[r] + bias, 0.0f);
  }
  __syncthreads();   // final layer reads rows produced by other waves

  // ---- layer 2: out = relu(y2 @ mW2 + mb2), one row per thread ----
  if (tid < 128) {
    float s = mb2[0];
    const float* yr = s_y2 + tid * 64;
#pragma unroll
    for (int k = 0; k < 64; ++k) s += yr[k] * mW2[k];   // mW2 reads are uniform -> scalarized
    out[row0 + tid] = fmaxf(s, 0.0f);
  }
}

// ---------------- launcher ---------------------------------------------------

extern "C" void kernel_launch(void* const* d_in, const int* in_sizes, int n_in,
                              void* d_out, int out_size, void* d_ws, size_t ws_size,
                              hipStream_t stream) {
  const int*   user_idx   = (const int*)  d_in[0];
  const int*   item_idx   = (const int*)  d_in[1];
  const int*   sample_ids = (const int*)  d_in[2];
  const int*   roots      = (const int*)  d_in[3];
  const float* user_infos = (const float*)d_in[4];
  const float* item_infos = (const float*)d_in[5];
  const float* adj_matrix = (const float*)d_in[6];
  const float* uid_emb    = (const float*)d_in[7];
  const float* sid_emb    = (const float*)d_in[8];
  const float* Wu         = (const float*)d_in[9];
  const float* bu         = (const float*)d_in[10];
  const float* Wi         = (const float*)d_in[11];
  const float* bi         = (const float*)d_in[12];
  const float* W1         = (const float*)d_in[13];
  const float* b1         = (const float*)d_in[14];
  const float* W2         = (const float*)d_in[15];
  const float* b2         = (const float*)d_in[16];
  const float* mW0        = (const float*)d_in[17];
  const float* mb0        = (const float*)d_in[18];
  const float* mW1        = (const float*)d_in[19];
  const float* mb1        = (const float*)d_in[20];
  const float* mW2        = (const float*)d_in[21];
  const float* mb2        = (const float*)d_in[22];
  float*       out        = (float*)d_out;

  // Workspace layout (bytes):
  //   adjx16  : NN*128*2        = 1,577,984
  //   W1p     : 65,536
  //   W2p     : 65,536
  //   mW0p    : 65,536
  //   mW1p    : 16,384
  //   xbuf    : BB*256*2        = 16,777,216
  char* ws = (char*)d_ws;
  size_t off = 0;
  _Float16* adjx16 = (_Float16*)(ws + off); off += (size_t)NN * DD * 2;
  _Float16* W1p    = (_Float16*)(ws + off); off += 128 * 256 * 2;
  _Float16* W2p    = (_Float16*)(ws + off); off += 256 * 128 * 2;
  _Float16* mW0p   = (_Float16*)(ws + off); off += 256 * 128 * 2;
  _Float16* mW1p   = (_Float16*)(ws + off); off += 128 * 64 * 2;
  _Float16* xbuf   = (_Float16*)(ws + off); off += (size_t)BB * 256 * 2;

  prep_adjx<<<NN, 128, 0, stream>>>(user_infos, item_infos, Wu, bu, Wi, bi, adjx16);
  pack_w<<<128, 256, 0, stream>>>(W1, W2, mW0, mW1, W1p, W2p, mW0p, mW1p);
  gnn_kernel<<<BB, 256, 0, stream>>>(user_idx, item_idx, sample_ids, roots,
                                     adj_matrix, uid_emb, sid_emb, adjx16,
                                     W1p, b1, W2p, b2, xbuf);
  mlp_kernel<<<BB / 128, 256, 0, stream>>>(xbuf, mW0p, mb0, mW1p, mb1, mW2, mb2, out);
}
